// SwitchMoELayer_37976100831590
// MI455X (gfx1250) — matrix operand order repaired and checked by reading, not compiled, not run on hardware
//
#include <hip/hip_runtime.h>
#include <hip/hip_bf16.h>

// Problem constants (from reference): B=16, T=4096, D=256, H=512, E=8
#define N_TOK 65536
#define D_IN  256
#define D_HID 512
#define N_EXP 8
#define TM    64          // tokens per expert tile
#define XPAD  8           // LDS row padding (bf16 elems) to dodge bank conflicts

typedef __attribute__((ext_vector_type(16))) __bf16 v16bf;
typedef __attribute__((ext_vector_type(8)))  __bf16 v8bf;
typedef __attribute__((ext_vector_type(8)))  float  v8f;

// ---------------------------------------------------------------- utilities
__global__ void zero_counts_kernel(int* counts) {
    if (threadIdx.x < 16) counts[threadIdx.x] = 0;
}

// Pack W1 [E][D][H] f32 -> bf16 in WMMA B-fragment order:
// idx = ((((e*32 + cblk)*8 + kblk)*32 + lane)*16 + i
// value = W1[e][kblk*32 + (lane>>4)*16 + i][cblk*16 + (lane&15)]
__global__ void pack_w1_kernel(const float* __restrict__ W1, __bf16* __restrict__ W1b) {
    const int idx  = blockIdx.x * blockDim.x + threadIdx.x;   // < 2^20 per... E*32*8*32*16 = 2^20
    const int i    = idx & 15;
    const int lane = (idx >> 4) & 31;
    const int kk   = (idx >> 9) & 7;
    const int cblk = (idx >> 12) & 31;
    const int e    = idx >> 17;
    const int K    = kk * 32 + (lane >> 4) * 16 + i;          // < 256
    const int col  = cblk * 16 + (lane & 15);                 // < 512
    W1b[idx] = (__bf16)W1[((size_t)e * D_IN + K) * D_HID + col];
}

// Pack W2 [E][H][D] f32 -> bf16, idx = ((((e*16 + cblk)*16 + kblk)*32 + lane)*16 + i
__global__ void pack_w2_kernel(const float* __restrict__ W2, __bf16* __restrict__ W2b) {
    const int idx  = blockIdx.x * blockDim.x + threadIdx.x;   // E*16*16*32*16 = 2^20
    const int i    = idx & 15;
    const int lane = (idx >> 4) & 31;
    const int kk   = (idx >> 9) & 15;
    const int cblk = (idx >> 13) & 15;
    const int e    = idx >> 17;
    const int K    = kk * 32 + (lane >> 4) * 16 + i;          // < 512
    const int col  = cblk * 16 + (lane & 15);                 // < 256
    W2b[idx] = (__bf16)W2[((size_t)e * D_HID + K) * D_IN + col];
}

// ---------------------------------------------------------------- gating
// logits = x @ gate_w + gate_b ; softmax ; top-2 ; append to (expert,slot) buckets
__global__ void __launch_bounds__(256)
gate_kernel(const float* __restrict__ x, const float* __restrict__ gw,
            const float* __restrict__ gb, int* __restrict__ counts,
            int* __restrict__ tokIdx, float* __restrict__ tokW) {
    __shared__ float sgw[D_IN * N_EXP];
    for (int i = threadIdx.x; i < D_IN * N_EXP; i += blockDim.x) sgw[i] = gw[i];
    __syncthreads();

    const int n = blockIdx.x * blockDim.x + threadIdx.x;
    float acc[N_EXP];
#pragma unroll
    for (int e = 0; e < N_EXP; ++e) acc[e] = gb[e];
    const float4* xr = (const float4*)(x + (size_t)n * D_IN);
    for (int j = 0; j < D_IN / 4; ++j) {
        const float4 v = xr[j];
        const float* g = &sgw[j * 4 * N_EXP];
#pragma unroll
        for (int e = 0; e < N_EXP; ++e)
            acc[e] += v.x * g[e] + v.y * g[N_EXP + e] + v.z * g[2 * N_EXP + e] + v.w * g[3 * N_EXP + e];
    }
    float mx = acc[0];
#pragma unroll
    for (int e = 1; e < N_EXP; ++e) mx = fmaxf(mx, acc[e]);
    float p[N_EXP], s = 0.f;
#pragma unroll
    for (int e = 0; e < N_EXP; ++e) { p[e] = expf(acc[e] - mx); s += p[e]; }
    const float inv = 1.0f / s;

    int e0 = 0; float v0 = p[0];
#pragma unroll
    for (int e = 1; e < N_EXP; ++e) if (p[e] > v0) { v0 = p[e]; e0 = e; }
    int e1 = -1; float v1 = -1.0f;
#pragma unroll
    for (int e = 0; e < N_EXP; ++e) if (e != e0 && p[e] > v1) { v1 = p[e]; e1 = e; }
    v0 *= inv; v1 *= inv;

    const int b0 = e0 * 2;                       // slot-0 bucket
    const int pos0 = atomicAdd(&counts[b0], 1);
    tokIdx[(size_t)b0 * N_TOK + pos0] = n;
    tokW [(size_t)b0 * N_TOK + pos0] = v0;
    const int b1 = e1 * 2 + 1;                   // slot-1 bucket
    const int pos1 = atomicAdd(&counts[b1], 1);
    tokIdx[(size_t)b1 * N_TOK + pos1] = n;
    tokW [(size_t)b1 * N_TOK + pos1] = v1;
}

// ---------------------------------------------------------------- expert FFN
// One block = 64 tokens of one (expert,slot) bucket.
// GEMM1 (64x256 @ 256x512) -> bias -> exact GELU -> bf16 LDS
// GEMM2 (64x512 @ 512x256) -> bias -> *combine weight -> scatter to out
__global__ void __launch_bounds__(256)
moe_expert_kernel(const int* __restrict__ counts, const int* __restrict__ tokIdx,
                  const float* __restrict__ tokW, const float* __restrict__ x,
                  const __bf16* __restrict__ W1b, const float* __restrict__ b1,
                  const __bf16* __restrict__ W2b, const float* __restrict__ b2,
                  float* __restrict__ out, const int slot) {
    __shared__ __bf16 Xs[TM][D_IN + XPAD];
    __shared__ __bf16 Hs[TM][D_HID + XPAD];
    __shared__ int    sTok[TM];
    __shared__ float  sWt[TM];

    const int e     = blockIdx.y;
    const int b     = e * 2 + slot;
    const int cnt   = counts[b];
    const int tile0 = blockIdx.x * TM;
    if (tile0 >= cnt) return;

    const int tid = threadIdx.x;
    if (tid < TM) {
        const int r  = tid;
        const int ok = (tile0 + r < cnt);
        const size_t src = (size_t)b * N_TOK + tile0 + (ok ? r : 0);
        sTok[r] = tokIdx[src];
        sWt[r]  = ok ? tokW[src] : 0.0f;
    }
    __syncthreads();

    // gather X tile (f32 -> bf16): 4 threads per row, 16 float4 each
    {
        const int r  = tid >> 2;
        const int cp = tid & 3;
        const float4* src = (const float4*)(x + (size_t)sTok[r] * D_IN);
#pragma unroll
        for (int j = 0; j < 16; ++j) {
            const int c4 = cp * 16 + j;
            const float4 v = src[c4];
            __bf16* dst = &Xs[r][c4 * 4];
            dst[0] = (__bf16)v.x; dst[1] = (__bf16)v.y;
            dst[2] = (__bf16)v.z; dst[3] = (__bf16)v.w;
        }
    }
    __syncthreads();

    const int wave   = tid >> 5;
    const int lane   = tid & 31;
    const int rg     = wave & 3;    // 16-row group
    const int colset = wave >> 2;   // column half
    const int m      = lane & 15;
    const int hi     = lane >> 4;

    // ---- GEMM1: Hs = gelu(X @ W1 + b1) ----
    for (int cb = 0; cb < 4; ++cb) {
        const int colbase = colset * 256 + cb * 64;
        v8f acc[4] = {};
        for (int kk = 0; kk < 8; ++kk) {
            // A frag: contiguous k order — lane(m,hi): K = hi*16+0..15 (matches packed B)
            const v8bf a0 = *(const v8bf*)&Xs[rg * 16 + m][kk * 32 + hi * 16];
            const v8bf a1 = *(const v8bf*)&Xs[rg * 16 + m][kk * 32 + hi * 16 + 8];
            const v16bf a = __builtin_shufflevector(a0, a1,
                0, 1, 2, 3, 4, 5, 6, 7, 8, 9, 10, 11, 12, 13, 14, 15);
#pragma unroll
            for (int t = 0; t < 4; ++t) {
                const int cblk = (colbase >> 4) + t;
                const v16bf bf = *(const v16bf*)
                    &W1b[((((size_t)e * 32 + cblk) * 8 + kk) * 32 + lane) * 16];
                acc[t] = __builtin_amdgcn_wmma_f32_16x16x32_bf16(
                    false, a, false, bf, (short)0, acc[t], false, false);
            }
        }
#pragma unroll
        for (int t = 0; t < 4; ++t) {
            const int col = colbase + t * 16 + m;
            const float bias = b1[e * D_HID + col];
#pragma unroll
            for (int r = 0; r < 8; ++r) {
                float v = acc[t][r] + bias;
                v = 0.5f * v * (1.0f + erff(v * 0.70710678118654752f));  // exact GELU
                Hs[rg * 16 + hi * 8 + r][col] = (__bf16)v;
            }
        }
    }
    __syncthreads();

    // ---- GEMM2: out[token] (or +=) = wt * (H @ W2 + b2) ----
    for (int cb = 0; cb < 2; ++cb) {
        const int colbase = colset * 128 + cb * 64;
        v8f acc[4] = {};
        for (int kk = 0; kk < 16; ++kk) {
            const v8bf a0 = *(const v8bf*)&Hs[rg * 16 + m][kk * 32 + hi * 16];
            const v8bf a1 = *(const v8bf*)&Hs[rg * 16 + m][kk * 32 + hi * 16 + 8];
            const v16bf a = __builtin_shufflevector(a0, a1,
                0, 1, 2, 3, 4, 5, 6, 7, 8, 9, 10, 11, 12, 13, 14, 15);
#pragma unroll
            for (int t = 0; t < 4; ++t) {
                const int cblk = (colbase >> 4) + t;
                const v16bf bf = *(const v16bf*)
                    &W2b[((((size_t)e * 16 + cblk) * 16 + kk) * 32 + lane) * 16];
                acc[t] = __builtin_amdgcn_wmma_f32_16x16x32_bf16(
                    false, a, false, bf, (short)0, acc[t], false, false);
            }
        }
#pragma unroll
        for (int t = 0; t < 4; ++t) {
            const int col = colbase + t * 16 + m;
            const float bias = b2[e * D_IN + col];
#pragma unroll
            for (int r = 0; r < 8; ++r) {
                const int row = rg * 16 + hi * 8 + r;
                if (tile0 + row < cnt) {
                    const float v = sWt[row] * (acc[t][r] + bias);
                    float* p = out + (size_t)sTok[row] * D_IN + col;
                    if (slot == 0) *p = v;      // pass 1: every token written once
                    else           *p += v;     // pass 2: ordered after pass 1
                }
            }
        }
    }
}

// ---------------------------------------------------------------- launcher
extern "C" void kernel_launch(void* const* d_in, const int* in_sizes, int n_in,
                              void* d_out, int out_size, void* d_ws, size_t ws_size,
                              hipStream_t stream) {
    const float* x  = (const float*)d_in[0];
    const float* gw = (const float*)d_in[1];
    const float* gb = (const float*)d_in[2];
    const float* W1 = (const float*)d_in[3];
    const float* b1 = (const float*)d_in[4];
    const float* W2 = (const float*)d_in[5];
    const float* b2 = (const float*)d_in[6];
    float* out = (float*)d_out;

    char* ws = (char*)d_ws;
    int*    counts = (int*)ws;                                        // 16 ints
    int*    tokIdx = (int*)(ws + 256);                                // 16*N ints (4 MB)
    float*  tokW   = (float*)(ws + 256 + (size_t)16 * N_TOK * 4);     // 16*N f32 (4 MB)
    __bf16* W1b    = (__bf16*)(ws + 256 + (size_t)16 * N_TOK * 8);    // 2 MB
    __bf16* W2b    = W1b + (size_t)N_EXP * D_IN * D_HID;              // 2 MB

    zero_counts_kernel<<<1, 32, 0, stream>>>(counts);
    pack_w1_kernel<<<(N_EXP * 32 * 8 * 32 * 16) / 256, 256, 0, stream>>>(W1, W1b);
    pack_w2_kernel<<<(N_EXP * 16 * 16 * 32 * 16) / 256, 256, 0, stream>>>(W2, W2b);
    gate_kernel<<<N_TOK / 256, 256, 0, stream>>>(x, gw, gb, counts, tokIdx, tokW);

    dim3 grid(N_TOK / TM, N_EXP);
    moe_expert_kernel<<<grid, 256, 0, stream>>>(counts, tokIdx, tokW, x, W1b, b1, W2b, b2, out, 0);
    moe_expert_kernel<<<grid, 256, 0, stream>>>(counts, tokIdx, tokW, x, W1b, b1, W2b, b2, out, 1);
}
